// OGSFNet_74818330296883
// MI455X (gfx1250) — compile-verified, matrix operand order checked by reading, not run on hardware
//
#include <hip/hip_runtime.h>
#include <hip/hip_bf16.h>
#include <math.h>

typedef __attribute__((ext_vector_type(16))) _Float16 v16h;
typedef __attribute__((ext_vector_type(8)))  float    v8f;

#define ACT_NONE  0
#define ACT_RELU  1
#define ACT_LRELU 2
#define ACT_SIG   3
#define ACT_CLIP  4

__device__ __forceinline__ float apply_act(float v, int act) {
  switch (act) {
    case ACT_RELU:  return v > 0.f ? v : 0.f;
    case ACT_LRELU: return v > 0.f ? v : 0.1f * v;
    case ACT_SIG:   return 1.f / (1.f + expf(-v));
    case ACT_CLIP:  return fminf(fmaxf(v, -200.f), 200.f);
    default:        return v;
  }
}

union Frag16 { v16h h; unsigned u[8]; };
union H2     { unsigned u; _Float16 h[2]; };

// ---------------------------------------------------------------------------
// WMMA GEMM: Y[n,cout] = act(X[n,cin] @ W[cin,cout] + b)
// One wave -> one 16x16 output tile; 4 waves/block.
// A tile in LDS as f16 rows; B tile in LDS as K-pair-packed dwords so both
// fragments are built with 8x ds_load_b32 (no u16 gathers / repacking).
// ---------------------------------------------------------------------------
__global__ __launch_bounds__(128) void k_gemm(
    const float* __restrict__ X, const float* __restrict__ W,
    const float* __restrict__ bias, float* __restrict__ Y,
    int n, int cin, int cout, int act)
{
  __shared__ _Float16 sA[4][16][32];     // [row][k]
  __shared__ unsigned sBp[4][16][16];    // [kpair][col] -> halves (2kp, 2kp+1)
  const int wave = threadIdx.x >> 5;
  const int lane = threadIdx.x & 31;
  const int half = (lane >> 4) & 1;
  const int l    = lane & 15;
  const int row0 = (blockIdx.x * 4 + wave) * 16;
  const int col0 = blockIdx.y * 16;
  const int ar  = lane >> 1;          // A row staged by this lane (0..15)
  const int acb = (lane & 1) * 16;    // A col base (0 or 16)
  const int bp  = lane & 15;          // B k-pair staged by this lane (0..15)
  const int bh  = (lane >> 4) * 8;    // B col base (0 or 8)
  const bool colFull = (col0 + 16 <= cout);
  const bool rowFull = (row0 + 16 <= n);

  v8f acc = {};
  for (int kk = 0; kk < cin; kk += 32) {
    // ---- stage A tile (16 x 32) ----
    {
      const float* Xp = X + (size_t)(row0 + ar) * cin + kk + acb;
      _Float16 t16[16];
      if (rowFull && (kk + 32 <= cin)) {
#pragma unroll
        for (int t = 0; t < 16; ++t) t16[t] = (_Float16)Xp[t];
      } else {
#pragma unroll
        for (int t = 0; t < 16; ++t) {
          float v = (row0 + ar < n && kk + acb + t < cin) ? Xp[t] : 0.f;
          t16[t] = (_Float16)v;
        }
      }
#pragma unroll
      for (int t = 0; t < 16; ++t) sA[wave][ar][acb + t] = t16[t];
    }
    // ---- stage B tile (32 x 16) as packed K-pairs ----
    {
      const float* W0 = W + (size_t)(kk + 2 * bp) * cout + col0 + bh;
      const float* W1 = W0 + cout;
      unsigned pk[8];
      if (colFull && (kk + 32 <= cin)) {
#pragma unroll
        for (int t = 0; t < 8; ++t) {
          H2 x; x.h[0] = (_Float16)W0[t]; x.h[1] = (_Float16)W1[t];
          pk[t] = x.u;
        }
      } else {
#pragma unroll
        for (int t = 0; t < 8; ++t) {
          bool cok = (col0 + bh + t < cout);
          float v0 = (kk + 2 * bp     < cin && cok) ? W0[t] : 0.f;
          float v1 = (kk + 2 * bp + 1 < cin && cok) ? W1[t] : 0.f;
          H2 x; x.h[0] = (_Float16)v0; x.h[1] = (_Float16)v1;
          pk[t] = x.u;
        }
      }
#pragma unroll
      for (int t = 0; t < 8; ++t) sBp[wave][bp][bh + t] = pk[t];
    }
    __syncthreads();
    // ---- fragments: 8x ds_load_b32 each ----
    Frag16 fa, fb;
#pragma unroll
    for (int v = 0; v < 8; ++v) {
      int kbase = ((v < 4) ? (v * 2) : (16 + (v - 4) * 2)) + half * 8;
      fa.u[v] = *(const unsigned*)&sA[wave][l][kbase];   // 4B-aligned K-pair
      int kp = ((v < 4) ? v : (8 + (v - 4))) + half * 4; // kbase / 2
      fb.u[v] = sBp[wave][kp][l];
    }
    acc = __builtin_amdgcn_wmma_f32_16x16x32_f16(false, fa.h, false, fb.h,
                                                 (short)0, acc, false, false);
    __syncthreads();
  }
  // ---- epilogue ----
  if (rowFull && colFull) {
#pragma unroll
    for (int j = 0; j < 8; ++j) {
      int r  = row0 + j + half * 8;
      int cc = col0 + l;
      Y[(size_t)r * cout + cc] = apply_act(acc[j] + bias[cc], act);
    }
  } else {
#pragma unroll
    for (int j = 0; j < 8; ++j) {
      int r  = row0 + j + half * 8;
      int cc = col0 + l;
      if (r < n && cc < cout)
        Y[(size_t)r * cout + cc] = apply_act(acc[j] + bias[cc], act);
    }
  }
}

// ---------------------------------------------------------------------------
// WMMA per-point aggregation: agg[s, ch*16 + w] = sum_k gp[s,k,ch] * wg[s,k,w]
// One wave per (point, 16-channel tile); K (template: 9 or 16) zero-padded
// to 32 -- the K-predicate constant-folds after unrolling.
// ---------------------------------------------------------------------------
template <int K>
__global__ __launch_bounds__(128) void k_pcagg(
    const float* __restrict__ gp, const float* __restrict__ wg,
    float* __restrict__ agg, int n, int C)
{
  const int wave = threadIdx.x >> 5;
  const int lane = threadIdx.x & 31;
  const int half = (lane >> 4) & 1;
  const int l    = lane & 15;
  const int ct   = (C + 15) >> 4;
  const int gid  = blockIdx.x * 4 + wave;
  const int s    = gid / ct;
  if (s >= n) return;
  const int c0 = (gid % ct) * 16;
  const bool fullC = (c0 + 16 <= C);
  const float* gpr = gp + (size_t)s * K * C + c0 + l;   // + kc*C per element
  const float* wgr = wg + (size_t)s * K * 16 + l;       // + kc*16 per element

  v16h a = {}, b = {};
#pragma unroll
  for (int i = 0; i < 16; ++i) {
    const int v    = i >> 1;
    const int base = ((v < 4) ? (v * 2) : (16 + (v - 4) * 2)) + (i & 1);
    const int kidx = base + half * 8;
    bool inK;
    if (base + 8 < K)      inK = true;          // both halves in range
    else if (base >= K)    inK = false;         // both halves out of range
    else                   inK = (kidx < K);    // lane-dependent middle zone
    const int kc = inK ? kidx : 0;              // clamped, always in-bounds
    float bv = wgr[(size_t)kc * 16];
    b[i] = (_Float16)(inK ? bv : 0.f);
    float av;
    if (fullC) av = gpr[(size_t)kc * C];
    else       av = (c0 + l < C) ? gpr[(size_t)kc * C] : 0.f;
    a[i] = (_Float16)(inK ? av : 0.f);
  }
  v8f acc = {};
  acc = __builtin_amdgcn_wmma_f32_16x16x32_f16(false, a, false, b,
                                               (short)0, acc, false, false);
  float* outp = agg + (size_t)s * C * 16 + (size_t)c0 * 16 + l;
  if (fullC) {
#pragma unroll
    for (int j = 0; j < 8; ++j)
      outp[(size_t)(j + half * 8) * 16] = acc[j];
  } else {
#pragma unroll
    for (int j = 0; j < 8; ++j) {
      int ch = c0 + j + half * 8;
      if (ch < C) agg[(size_t)s * C * 16 + (size_t)ch * 16 + l] = acc[j];
    }
  }
}

// ---------------------------------------------------------------------------
// Small dense (cout < 16): one thread per row.
// ---------------------------------------------------------------------------
__global__ void k_dense_small(const float* __restrict__ X, const float* __restrict__ W,
                              const float* __restrict__ B, float* __restrict__ Y,
                              int n, int cin, int cout, int act)
{
  int r = blockIdx.x * blockDim.x + threadIdx.x;
  if (r >= n) return;
  for (int c = 0; c < cout; ++c) {
    float acc = B[c];
    for (int i = 0; i < cin; ++i) acc += X[(size_t)r * cin + i] * W[(size_t)i * cout + c];
    Y[(size_t)r * cout + c] = apply_act(acc, act);
  }
}

// ---------------------------------------------------------------------------
// Fused WeightNet: 3 -> 8 -> 8 -> cout, ReLU after every layer. Weights in LDS.
// ---------------------------------------------------------------------------
__global__ __launch_bounds__(128) void k_weightnet(
    const float* __restrict__ gx,
    const float* __restrict__ w1, const float* __restrict__ b1,
    const float* __restrict__ w2, const float* __restrict__ b2,
    const float* __restrict__ w3, const float* __restrict__ b3,
    float* __restrict__ out, int rows, int cout)
{
  __shared__ float sP[104];      // w1[24] b1[8] w2[64] b2[8]
  __shared__ float sW3[8 * 256];
  __shared__ float sB3[256];
  for (int i = threadIdx.x; i < 24; i += blockDim.x) sP[i] = w1[i];
  for (int i = threadIdx.x; i < 8;  i += blockDim.x) sP[24 + i] = b1[i];
  for (int i = threadIdx.x; i < 64; i += blockDim.x) sP[32 + i] = w2[i];
  for (int i = threadIdx.x; i < 8;  i += blockDim.x) sP[96 + i] = b2[i];
  for (int i = threadIdx.x; i < 8 * cout; i += blockDim.x) sW3[i] = w3[i];
  for (int i = threadIdx.x; i < cout; i += blockDim.x) sB3[i] = b3[i];
  __syncthreads();
  int r = blockIdx.x * blockDim.x + threadIdx.x;
  if (r >= rows) return;
  float x0 = gx[(size_t)r * 3], x1 = gx[(size_t)r * 3 + 1], x2 = gx[(size_t)r * 3 + 2];
  float h[8], g[8];
#pragma unroll
  for (int j = 0; j < 8; ++j) {
    float a = sP[24 + j] + x0 * sP[j] + x1 * sP[8 + j] + x2 * sP[16 + j];
    h[j] = a > 0.f ? a : 0.f;
  }
#pragma unroll
  for (int j = 0; j < 8; ++j) {
    float a = sP[96 + j];
#pragma unroll
    for (int i = 0; i < 8; ++i) a += h[i] * sP[32 + i * 8 + j];
    g[j] = a > 0.f ? a : 0.f;
  }
  for (int cc = 0; cc < cout; ++cc) {
    float a = sB3[cc];
#pragma unroll
    for (int i = 0; i < 8; ++i) a += g[i] * sW3[i * cout + cc];
    out[(size_t)r * cout + cc] = a > 0.f ? a : 0.f;
  }
}

// ---------------------------------------------------------------------------
// Brute-force KNN with LDS-tiled references and register top-K.
// ---------------------------------------------------------------------------
template <int K>
__global__ __launch_bounds__(256) void k_knn(const float* __restrict__ q, int n,
                                             const float* __restrict__ ref, int m,
                                             int* __restrict__ idx)
{
  __shared__ float sx[256], sy[256], sz[256];
  int t = blockIdx.x * 256 + threadIdx.x;
  bool active = t < n;
  float qx = 0.f, qy = 0.f, qz = 0.f;
  if (active) { qx = q[(size_t)t * 3]; qy = q[(size_t)t * 3 + 1]; qz = q[(size_t)t * 3 + 2]; }
  float bd[K]; int bi[K];
#pragma unroll
  for (int i = 0; i < K; ++i) { bd[i] = 3.4e38f; bi[i] = 0; }
  for (int base = 0; base < m; base += 256) {
    int j = base + threadIdx.x;
    if (j < m) {
      sx[threadIdx.x] = ref[(size_t)j * 3];
      sy[threadIdx.x] = ref[(size_t)j * 3 + 1];
      sz[threadIdx.x] = ref[(size_t)j * 3 + 2];
    }
    __syncthreads();
    int lim = m - base; if (lim > 256) lim = 256;
    if (active) {
      for (int u = 0; u < lim; ++u) {
        float dx = qx - sx[u], dy = qy - sy[u], dz = qz - sz[u];
        float d = dx * dx + dy * dy + dz * dz;
        if (d < bd[K - 1]) {
          bd[K - 1] = d; bi[K - 1] = base + u;
#pragma unroll
          for (int s = K - 1; s > 0; --s) {
            if (bd[s] < bd[s - 1]) {
              float td = bd[s]; bd[s] = bd[s - 1]; bd[s - 1] = td;
              int ti = bi[s]; bi[s] = bi[s - 1]; bi[s - 1] = ti;
            } else break;
          }
        }
      }
    }
    __syncthreads();
  }
  if (active) {
#pragma unroll
    for (int i = 0; i < K; ++i) idx[(size_t)t * K + i] = bi[i];
  }
}

// ---------------------------------------------------------------------------
// Farthest point sampling: single workgroup, LDS argmax reduction.
// ---------------------------------------------------------------------------
__global__ __launch_bounds__(1024) void k_fps(const float* __restrict__ xyz, int n,
                                              int npoint, int* __restrict__ out,
                                              float* __restrict__ dist)
{
  __shared__ float sred[1024];
  __shared__ int   sidx[1024];
  __shared__ float cen[3];
  __shared__ int   curs;
  int t = threadIdx.x;
  for (int i = t; i < n; i += 1024) dist[i] = 1e10f;
  if (t == 0) curs = 0;
  __syncthreads();
  for (int it = 0; it < npoint; ++it) {
    if (t == 0) {
      out[it] = curs;
      cen[0] = xyz[(size_t)curs * 3];
      cen[1] = xyz[(size_t)curs * 3 + 1];
      cen[2] = xyz[(size_t)curs * 3 + 2];
    }
    __syncthreads();
    float best = -1.f; int besti = 0;
    for (int i = t; i < n; i += 1024) {
      float dx = xyz[(size_t)i * 3] - cen[0];
      float dy = xyz[(size_t)i * 3 + 1] - cen[1];
      float dz = xyz[(size_t)i * 3 + 2] - cen[2];
      float d = dx * dx + dy * dy + dz * dz;
      float nd = fminf(dist[i], d);
      dist[i] = nd;
      if (nd > best) { best = nd; besti = i; }
    }
    sred[t] = best; sidx[t] = besti;
    __syncthreads();
    for (int s = 512; s > 0; s >>= 1) {
      if (t < s && sred[t + s] > sred[t]) { sred[t] = sred[t + s]; sidx[t] = sidx[t + s]; }
      __syncthreads();
    }
    if (t == 0) curs = sidx[0];
    __syncthreads();
  }
}

// ---------------------------------------------------------------------------
// Gathers / builders / reductions / elementwise.
// ---------------------------------------------------------------------------
__global__ void k_gather_rows(const float* __restrict__ src, const int* __restrict__ idx,
                              float* __restrict__ out, int n, int C)
{
  int e = blockIdx.x * blockDim.x + threadIdx.x;
  if (e >= n * C) return;
  int r = e / C, c = e % C;
  out[e] = src[(size_t)idx[r] * C + c];
}

__global__ void k_gx(const float* __restrict__ xyz, const int* __restrict__ idx,
                     const float* __restrict__ cen, float* __restrict__ gx,
                     int n, int k)
{
  int e = blockIdx.x * blockDim.x + threadIdx.x;
  if (e >= n * k * 3) return;
  int c = e % 3, r = e / 3, qn = r / k;
  gx[e] = xyz[(size_t)idx[r] * 3 + c] - cen[(size_t)qn * 3 + c];
}

__global__ void k_build_gp(const float* __restrict__ gx, const float* __restrict__ feats,
                           const int* __restrict__ idx, float* __restrict__ gp,
                           int n, int k, int C)
{
  int Cg = C + 3;
  int e = blockIdx.x * blockDim.x + threadIdx.x;
  if (e >= n * k * Cg) return;
  int r = e / Cg, c = e % Cg;
  gp[e] = (c < 3) ? gx[(size_t)r * 3 + c] : feats[(size_t)idx[r] * C + (c - 3)];
}

__global__ void k_build_cvx(const float* __restrict__ f1, const float* __restrict__ f2,
                            const int* __restrict__ idx, const float* __restrict__ dirv,
                            float* __restrict__ out, int n, int k, int C1, int C2)
{
  int Ct = C1 + C2 + 3;
  int e = blockIdx.x * blockDim.x + threadIdx.x;
  if (e >= n * k * Ct) return;
  int r = e / Ct, c = e % Ct, qn = r / k;
  float v;
  if (c < C1)           v = f1[(size_t)qn * C1 + c];
  else if (c < C1 + C2) v = f2[(size_t)idx[r] * C2 + (c - C1)];
  else                  v = dirv[(size_t)r * 3 + (c - C1 - C2)];
  out[e] = v;
}

__global__ void k_concat4(const float* __restrict__ a, int Ca,
                          const float* __restrict__ b, int Cb,
                          const float* __restrict__ cc, int Cc,
                          const float* __restrict__ d, int Cd,
                          float* __restrict__ out, int n)
{
  int Ct = Ca + Cb + Cc + Cd;
  int e = blockIdx.x * blockDim.x + threadIdx.x;
  if (e >= n * Ct) return;
  int r = e / Ct, ch = e % Ct;
  float v;
  if (ch < Ca)                v = a[(size_t)r * Ca + ch];
  else if (ch < Ca + Cb)      v = b[(size_t)r * Cb + ch - Ca];
  else if (ch < Ca + Cb + Cc) v = cc[(size_t)r * Cc + ch - Ca - Cb];
  else                        v = d[(size_t)r * Cd + ch - Ca - Cb - Cc];
  out[e] = v;
}

__global__ void k_wsum(const float* __restrict__ w, const float* __restrict__ x,
                       float* __restrict__ out, int n, int k, int C)
{
  int e = blockIdx.x * blockDim.x + threadIdx.x;
  if (e >= n * C) return;
  int r = e / C, c = e % C;
  float acc = 0.f;
  for (int j = 0; j < k; ++j) {
    size_t o = ((size_t)r * k + j) * C + c;
    acc += w[o] * x[o];
  }
  out[e] = acc;
}

__global__ void k_wsum_gather(const float* __restrict__ w, const float* __restrict__ p2p,
                              const int* __restrict__ idx, float* __restrict__ out,
                              int n, int k, int C)
{
  int e = blockIdx.x * blockDim.x + threadIdx.x;
  if (e >= n * C) return;
  int r = e / C, c = e % C;
  float acc = 0.f;
  for (int j = 0; j < k; ++j)
    acc += w[((size_t)r * k + j) * C + c] * p2p[(size_t)idx[(size_t)r * k + j] * C + c];
  out[e] = acc;
}

__global__ void k_maskmul(float* __restrict__ p, const float* __restrict__ mask, int n, int C)
{
  int e = blockIdx.x * blockDim.x + threadIdx.x;
  if (e >= n * C) return;
  p[e] *= mask[e / C];
}

__global__ void k_interp(const float* __restrict__ q, const float* __restrict__ src,
                         const int* __restrict__ idx, const float* __restrict__ feat,
                         float* __restrict__ out, int n, int C)
{
  int r = blockIdx.x * blockDim.x + threadIdx.x;
  if (r >= n) return;
  int id[3]; float w[3]; float wsum = 0.f;
#pragma unroll
  for (int j = 0; j < 3; ++j) {
    id[j] = idx[(size_t)r * 3 + j];
    float dx = src[(size_t)id[j] * 3]     - q[(size_t)r * 3];
    float dy = src[(size_t)id[j] * 3 + 1] - q[(size_t)r * 3 + 1];
    float dz = src[(size_t)id[j] * 3 + 2] - q[(size_t)r * 3 + 2];
    float d = fmaxf(sqrtf(dx * dx + dy * dy + dz * dz), 1e-10f);
    w[j] = 1.f / d; wsum += w[j];
  }
#pragma unroll
  for (int j = 0; j < 3; ++j) w[j] /= wsum;
  for (int c = 0; c < C; ++c)
    out[(size_t)r * C + c] = w[0] * feat[(size_t)id[0] * C + c]
                           + w[1] * feat[(size_t)id[1] * C + c]
                           + w[2] * feat[(size_t)id[2] * C + c];
}

__global__ void k_add(const float* a, const float* b, float* o, int n)
{ int e = blockIdx.x * blockDim.x + threadIdx.x; if (e < n) o[e] = a[e] + b[e]; }
__global__ void k_sub(const float* a, const float* b, float* o, int n)
{ int e = blockIdx.x * blockDim.x + threadIdx.x; if (e < n) o[e] = a[e] - b[e]; }

// ===========================================================================
// Host-side orchestration
// ===========================================================================
struct DP  { const float* b; const float* w; };
struct PCP { DP lin; DP wn[3]; };
struct CVP { DP mlp[2]; DP wn1[3]; DP wn2[3]; };
struct SFP { DP fc, mlp1, mlp2, occ; PCP pc1, pc2; };

struct Ctx {
  hipStream_t st;
  char* ws;
  size_t lo, hi;
};

static inline size_t alup(size_t x) { return (x + 255) & ~(size_t)255; }
static float* permf(Ctx& c, size_t n) { float* p = (float*)(c.ws + c.lo); c.lo += alup(n * 4); return p; }
static float* tmpf(Ctx& c, size_t n)  { c.hi -= alup(n * 4); return (float*)(c.ws + c.hi); }
static int*   tmpi(Ctx& c, size_t n)  { c.hi -= alup(n * 4); return (int*)(c.ws + c.hi); }
static inline unsigned gE(size_t total) { return (unsigned)((total + 255) / 256); }

static void dense(Ctx& c, const float* X, int n, const DP& p, int cin, int cout,
                  float* Y, int act)
{
  if (cout >= 16) {
    dim3 g((unsigned)(((n + 15) / 16 + 3) / 4), (unsigned)((cout + 15) / 16));
    k_gemm<<<g, 128, 0, c.st>>>(X, p.w, p.b, Y, n, cin, cout, act);
  } else {
    k_dense_small<<<gE(n), 256, 0, c.st>>>(X, p.w, p.b, Y, n, cin, cout, act);
  }
}

static void pcagg(Ctx& c, const float* gp, const float* wg, float* agg,
                  int n, int k, int C)
{
  unsigned g = (unsigned)(((size_t)n * ((C + 15) / 16) + 3) / 4);
  if (k == 9) k_pcagg<9><<<g, 128, 0, c.st>>>(gp, wg, agg, n, C);
  else        k_pcagg<16><<<g, 128, 0, c.st>>>(gp, wg, agg, n, C);
}

static void weightnet(Ctx& c, const float* gx, int rows, const DP* wn, int cout, float* out)
{
  k_weightnet<<<(unsigned)((rows + 127) / 128), 128, 0, c.st>>>(
      gx, wn[0].w, wn[0].b, wn[1].w, wn[1].b, wn[2].w, wn[2].b, out, rows, cout);
}

static void knn(Ctx& c, const float* q, int n, const float* ref, int m, int k, int* idx)
{
  unsigned g = gE(n);
  switch (k) {
    case 3:  k_knn<3><<<g, 256, 0, c.st>>>(q, n, ref, m, idx);  break;
    case 9:  k_knn<9><<<g, 256, 0, c.st>>>(q, n, ref, m, idx);  break;
    case 16: k_knn<16><<<g, 256, 0, c.st>>>(q, n, ref, m, idx); break;
    default: k_knn<32><<<g, 256, 0, c.st>>>(q, n, ref, m, idx); break;
  }
}

static void pointconvd(Ctx& c, const PCP& p, const float* xyz, const float* feats,
                       int n, int C, int npoint, int k, int cout,
                       float* new_xyz, float* fout)
{
  size_t mk = c.hi;
  int* fi = tmpi(c, npoint);
  float* dist = tmpf(c, n);
  k_fps<<<1, 1024, 0, c.st>>>(xyz, n, npoint, fi, dist);
  k_gather_rows<<<gE((size_t)npoint * 3), 256, 0, c.st>>>(xyz, fi, new_xyz, npoint, 3);
  int* idx = tmpi(c, (size_t)npoint * k);
  knn(c, new_xyz, npoint, xyz, n, k, idx);
  float* gxb = tmpf(c, (size_t)npoint * k * 3);
  k_gx<<<gE((size_t)npoint * k * 3), 256, 0, c.st>>>(xyz, idx, new_xyz, gxb, npoint, k);
  int Cg = C + 3;
  float* gp = tmpf(c, (size_t)npoint * k * Cg);
  k_build_gp<<<gE((size_t)npoint * k * Cg), 256, 0, c.st>>>(gxb, feats, idx, gp, npoint, k, C);
  float* wg = tmpf(c, (size_t)npoint * k * 16);
  weightnet(c, gxb, npoint * k, p.wn, 16, wg);
  float* agg = tmpf(c, (size_t)npoint * Cg * 16);
  pcagg(c, gp, wg, agg, npoint, k, Cg);
  dense(c, agg, npoint, p.lin, Cg * 16, cout, fout, ACT_LRELU);
  c.hi = mk;
}

static void pointconv(Ctx& c, const PCP& p, const float* xyz, const float* feats,
                      int n, int C, int k, int cout, float* out)
{
  size_t mk = c.hi;
  int* idx = tmpi(c, (size_t)n * k);
  knn(c, xyz, n, xyz, n, k, idx);
  float* gxb = tmpf(c, (size_t)n * k * 3);
  k_gx<<<gE((size_t)n * k * 3), 256, 0, c.st>>>(xyz, idx, xyz, gxb, n, k);
  int Cg = C + 3;
  float* gp = tmpf(c, (size_t)n * k * Cg);
  k_build_gp<<<gE((size_t)n * k * Cg), 256, 0, c.st>>>(gxb, feats, idx, gp, n, k, C);
  float* wg = tmpf(c, (size_t)n * k * 16);
  weightnet(c, gxb, n * k, p.wn, 16, wg);
  float* agg = tmpf(c, (size_t)n * Cg * 16);
  pcagg(c, gp, wg, agg, n, k, Cg);
  dense(c, agg, n, p.lin, Cg * 16, cout, out, ACT_LRELU);
  c.hi = mk;
}

static void upsample(Ctx& c, const float* dq, int n, const float* ds, int m,
                     const float* sf, int C, float* out)
{
  size_t mk = c.hi;
  int* idx = tmpi(c, (size_t)n * 3);
  knn(c, dq, n, ds, m, 3, idx);
  k_interp<<<gE(n), 256, 0, c.st>>>(dq, ds, idx, sf, out, n, C);
  c.hi = mk;
}

static void warp(Ctx& c, const float* xyz1, int n, const float* xyz2, int m,
                 const float* flow1, float* out)
{
  size_t mk = c.hi;
  float* x12 = tmpf(c, (size_t)n * 3);
  k_add<<<gE((size_t)n * 3), 256, 0, c.st>>>(xyz1, flow1, x12, n * 3);
  int* idx = tmpi(c, (size_t)m * 3);
  knn(c, xyz2, m, x12, n, 3, idx);
  float* f2 = tmpf(c, (size_t)m * 3);
  k_interp<<<gE(m), 256, 0, c.st>>>(xyz2, x12, idx, flow1, f2, m, 3);
  k_sub<<<gE((size_t)m * 3), 256, 0, c.st>>>(xyz2, f2, out, m * 3);
  c.hi = mk;
}

static void mask_cv(Ctx& c, const CVP& p, const float* xyz1, int n,
                    const float* xyz2, int m, const float* f1, int C1,
                    const float* f2, int C2, const float* mask,
                    int c1ch, int c2ch, float* out)
{
  const int K = 32;
  size_t mk = c.hi;
  int* idx = tmpi(c, (size_t)n * K);
  knn(c, xyz1, n, xyz2, m, K, idx);
  float* dirv = tmpf(c, (size_t)n * K * 3);
  k_gx<<<gE((size_t)n * K * 3), 256, 0, c.st>>>(xyz2, idx, xyz1, dirv, n, K);
  int Cin = C1 + C2 + 3;
  float* x0 = tmpf(c, (size_t)n * K * Cin);
  k_build_cvx<<<gE((size_t)n * K * Cin), 256, 0, c.st>>>(f1, f2, idx, dirv, x0, n, K, C1, C2);
  float* x1 = tmpf(c, (size_t)n * K * c1ch);
  dense(c, x0, n * K, p.mlp[0], Cin, c1ch, x1, ACT_LRELU);
  float* x2 = tmpf(c, (size_t)n * K * c2ch);
  dense(c, x1, n * K, p.mlp[1], c1ch, c2ch, x2, ACT_LRELU);
  float* w1 = tmpf(c, (size_t)n * K * c2ch);
  weightnet(c, dirv, n * K, p.wn1, c2ch, w1);
  float* p2p = tmpf(c, (size_t)n * c2ch);
  k_wsum<<<gE((size_t)n * c2ch), 256, 0, c.st>>>(w1, x2, p2p, n, K, c2ch);
  if (mask) k_maskmul<<<gE((size_t)n * c2ch), 256, 0, c.st>>>(p2p, mask, n, c2ch);
  int* idx2 = tmpi(c, (size_t)n * K);
  knn(c, xyz1, n, xyz1, n, K, idx2);
  float* dir2 = tmpf(c, (size_t)n * K * 3);
  k_gx<<<gE((size_t)n * K * 3), 256, 0, c.st>>>(xyz1, idx2, xyz1, dir2, n, K);
  float* w2 = tmpf(c, (size_t)n * K * c2ch);
  weightnet(c, dir2, n * K, p.wn2, c2ch, w2);
  k_wsum_gather<<<gE((size_t)n * c2ch), 256, 0, c.st>>>(w2, p2p, idx2, out, n, K, c2ch);
  c.hi = mk;
}

static void sfoe(Ctx& c, const SFP& p, const float* xyz, int n,
                 const float* feats, int Cf, const float* cost, int Cc,
                 const float* flow, const float* occ,
                 float* feat_out, float* flow_out, float* occ_out)
{
  size_t mk = c.hi;
  int Cin = Cf + Cc + (flow ? 3 : 0) + (occ ? 1 : 0);
  float* x = tmpf(c, (size_t)n * Cin);
  k_concat4<<<gE((size_t)n * Cin), 256, 0, c.st>>>(
      feats, Cf, cost, Cc, flow, flow ? 3 : 0, occ, occ ? 1 : 0, x, n);
  float* x1 = tmpf(c, (size_t)n * 128);
  pointconv(c, p.pc1, xyz, x, n, Cin, 9, 128, x1);
  float* x2 = tmpf(c, (size_t)n * 128);
  pointconv(c, p.pc2, xyz, x1, n, 128, 9, 128, x2);
  float* x3 = tmpf(c, (size_t)n * 128);
  dense(c, x2, n, p.mlp1, 128, 128, x3, ACT_LRELU);
  dense(c, x3, n, p.mlp2, 128, 64, feat_out, ACT_LRELU);
  k_dense_small<<<gE(n), 256, 0, c.st>>>(feat_out, p.fc.w, p.fc.b, flow_out, n, 64, 3, ACT_CLIP);
  k_dense_small<<<gE(n), 256, 0, c.st>>>(feat_out, p.occ.w, p.occ.b, occ_out, n, 64, 1, ACT_SIG);
  c.hi = mk;
}

extern "C" void kernel_launch(void* const* d_in, const int* in_sizes, int n_in,
                              void* d_out, int out_size, void* d_ws, size_t ws_size,
                              hipStream_t stream)
{
  (void)in_sizes; (void)n_in; (void)out_size;
  const float* xyz1   = (const float*)d_in[0];
  const float* xyz2   = (const float*)d_in[1];
  const float* color1 = (const float*)d_in[2];
  const float* color2 = (const float*)d_in[3];

  // Params flattened in jax dict-sorted order; dp -> (b, w).
  int pi = 4;
  auto DPn  = [&]() { DP d; d.b = (const float*)d_in[pi++]; d.w = (const float*)d_in[pi++]; return d; };
  auto WN3  = [&](DP* w) { w[0] = DPn(); w[1] = DPn(); w[2] = DPn(); };
  auto PCPn = [&]() { PCP p; p.lin = DPn(); WN3(p.wn); return p; };
  auto CVPn = [&]() { CVP p; p.mlp[0] = DPn(); p.mlp[1] = DPn(); WN3(p.wn1); WN3(p.wn2); return p; };
  auto SFPn = [&]() { SFP p; p.fc = DPn(); p.mlp1 = DPn(); p.mlp2 = DPn(); p.occ = DPn();
                      p.pc1 = PCPn(); p.pc2 = PCPn(); return p; };

  CVP Pcost0 = CVPn(), Pcost1 = CVPn(), Pcost2 = CVPn(), Pcost3 = CVPn();
  DP  Pd10 = DPn(), Pd21 = DPn(), Pd32 = DPn(), Pd43 = DPn();
  SFP Pf0 = SFPn(), Pf1 = SFPn(), Pf2 = SFPn(), Pf3 = SFPn();
  DP  Pl0 = DPn(), Pl0_1 = DPn(), Pl0_2 = DPn();
  PCP Pl1 = PCPn(); DP Pl1_0 = DPn(), Pl1_1 = DPn();
  PCP Pl2 = PCPn(); DP Pl2_0 = DPn(), Pl2_1 = DPn();
  PCP Pl3 = PCPn(); DP Pl3_0 = DPn(), Pl3_1 = DPn();
  PCP Pl4 = PCPn();

  Ctx c; c.st = stream; c.ws = (char*)d_ws; c.lo = 0; c.hi = ws_size & ~(size_t)255;

  const int NL0 = 8192, NL1 = 2048, NL2 = 512, NL3 = 256, NL4 = 128;
  float* out = (float*)d_out;
  float* o_flow0 = out;          float* o_flow1 = out + 24576;
  float* o_flow2 = out + 30720;  float* o_flow3 = out + 32256;
  float* o_occ0  = out + 33024;  float* o_occ1  = out + 41216;
  float* o_occ2  = out + 43264;  float* o_occ3  = out + 43776;

  // Level-0 feature MLPs
  float* f1_0   = permf(c, (size_t)NL0 * 32);
  float* f1_0_1 = permf(c, (size_t)NL0 * 64);
  float* f2_0   = permf(c, (size_t)NL0 * 32);
  float* f2_0_1 = permf(c, (size_t)NL0 * 64);
  {
    size_t mk = c.hi;
    float* t = tmpf(c, (size_t)NL0 * 32);
    dense(c, color1, NL0, Pl0, 3, 32, t, ACT_LRELU);
    dense(c, t, NL0, Pl0_1, 32, 32, f1_0, ACT_LRELU);
    dense(c, f1_0, NL0, Pl0_2, 32, 64, f1_0_1, ACT_LRELU);
    dense(c, color2, NL0, Pl0, 3, 32, t, ACT_LRELU);
    dense(c, t, NL0, Pl0_1, 32, 32, f2_0, ACT_LRELU);
    dense(c, f2_0, NL0, Pl0_2, 32, 64, f2_0_1, ACT_LRELU);
    c.hi = mk;
  }

  // Pyramid level 1
  float* pc1_1 = permf(c, (size_t)NL1 * 3); float* f1_1 = permf(c, (size_t)NL1 * 64);
  pointconvd(c, Pl1, xyz1, f1_0_1, NL0, 64, NL1, 16, 64, pc1_1, f1_1);
  float* f1_1_2 = permf(c, (size_t)NL1 * 128);
  { size_t mk = c.hi; float* t = tmpf(c, (size_t)NL1 * 64);
    dense(c, f1_1, NL1, Pl1_0, 64, 64, t, ACT_LRELU);
    dense(c, t, NL1, Pl1_1, 64, 128, f1_1_2, ACT_LRELU); c.hi = mk; }
  float* pc2_1 = permf(c, (size_t)NL1 * 3); float* f2_1 = permf(c, (size_t)NL1 * 64);
  pointconvd(c, Pl1, xyz2, f2_0_1, NL0, 64, NL1, 16, 64, pc2_1, f2_1);
  float* f2_1_2 = permf(c, (size_t)NL1 * 128);
  { size_t mk = c.hi; float* t = tmpf(c, (size_t)NL1 * 64);
    dense(c, f2_1, NL1, Pl1_0, 64, 64, t, ACT_LRELU);
    dense(c, t, NL1, Pl1_1, 64, 128, f2_1_2, ACT_LRELU); c.hi = mk; }

  // Level 2
  float* pc1_2 = permf(c, (size_t)NL2 * 3); float* f1_2 = permf(c, (size_t)NL2 * 128);
  pointconvd(c, Pl2, pc1_1, f1_1_2, NL1, 128, NL2, 16, 128, pc1_2, f1_2);
  float* f1_2_3 = permf(c, (size_t)NL2 * 256);
  { size_t mk = c.hi; float* t = tmpf(c, (size_t)NL2 * 128);
    dense(c, f1_2, NL2, Pl2_0, 128, 128, t, ACT_LRELU);
    dense(c, t, NL2, Pl2_1, 128, 256, f1_2_3, ACT_LRELU); c.hi = mk; }
  float* pc2_2 = permf(c, (size_t)NL2 * 3); float* f2_2 = permf(c, (size_t)NL2 * 128);
  pointconvd(c, Pl2, pc2_1, f2_1_2, NL1, 128, NL2, 16, 128, pc2_2, f2_2);
  float* f2_2_3 = permf(c, (size_t)NL2 * 256);
  { size_t mk = c.hi; float* t = tmpf(c, (size_t)NL2 * 128);
    dense(c, f2_2, NL2, Pl2_0, 128, 128, t, ACT_LRELU);
    dense(c, t, NL2, Pl2_1, 128, 256, f2_2_3, ACT_LRELU); c.hi = mk; }

  // Level 3
  float* pc1_3 = permf(c, (size_t)NL3 * 3); float* f1_3 = permf(c, (size_t)NL3 * 256);
  pointconvd(c, Pl3, pc1_2, f1_2_3, NL2, 256, NL3, 16, 256, pc1_3, f1_3);
  float* f1_3_4 = permf(c, (size_t)NL3 * 512);
  { size_t mk = c.hi; float* t = tmpf(c, (size_t)NL3 * 256);
    dense(c, f1_3, NL3, Pl3_0, 256, 256, t, ACT_LRELU);
    dense(c, t, NL3, Pl3_1, 256, 512, f1_3_4, ACT_LRELU); c.hi = mk; }
  float* pc2_3 = permf(c, (size_t)NL3 * 3); float* f2_3 = permf(c, (size_t)NL3 * 256);
  pointconvd(c, Pl3, pc2_2, f2_2_3, NL2, 256, NL3, 16, 256, pc2_3, f2_3);
  float* f2_3_4 = permf(c, (size_t)NL3 * 512);
  { size_t mk = c.hi; float* t = tmpf(c, (size_t)NL3 * 256);
    dense(c, f2_3, NL3, Pl3_0, 256, 256, t, ACT_LRELU);
    dense(c, t, NL3, Pl3_1, 256, 512, f2_3_4, ACT_LRELU); c.hi = mk; }

  // Level 4 + deconv4_3
  float* pc1_4 = permf(c, (size_t)NL4 * 3); float* f1_4 = permf(c, (size_t)NL4 * 256);
  pointconvd(c, Pl4, pc1_3, f1_3_4, NL3, 512, NL4, 16, 256, pc1_4, f1_4);
  float* f1_4_3 = permf(c, (size_t)NL3 * 64);
  { size_t mk = c.hi; float* u = tmpf(c, (size_t)NL3 * 256);
    upsample(c, pc1_3, NL3, pc1_4, NL4, f1_4, 256, u);
    dense(c, u, NL3, Pd43, 256, 64, f1_4_3, ACT_LRELU); c.hi = mk; }
  float* pc2_4 = permf(c, (size_t)NL4 * 3); float* f2_4 = permf(c, (size_t)NL4 * 256);
  pointconvd(c, Pl4, pc2_3, f2_3_4, NL3, 512, NL4, 16, 256, pc2_4, f2_4);
  float* f2_4_3 = permf(c, (size_t)NL3 * 64);
  { size_t mk = c.hi; float* u = tmpf(c, (size_t)NL3 * 256);
    upsample(c, pc2_3, NL3, pc2_4, NL4, f2_4, 256, u);
    dense(c, u, NL3, Pd43, 256, 64, f2_4_3, ACT_LRELU); c.hi = mk; }

  // Level-3 cost + flow
  float* c_f1_3 = permf(c, (size_t)NL3 * 320);
  k_concat4<<<gE((size_t)NL3 * 320), 256, 0, stream>>>(f1_3, 256, f1_4_3, 64, nullptr, 0, nullptr, 0, c_f1_3, NL3);
  float* c_f2_3 = permf(c, (size_t)NL3 * 320);
  k_concat4<<<gE((size_t)NL3 * 320), 256, 0, stream>>>(f2_3, 256, f2_4_3, 64, nullptr, 0, nullptr, 0, c_f2_3, NL3);
  float* cost3 = permf(c, (size_t)NL3 * 256);
  mask_cv(c, Pcost3, pc1_3, NL3, pc2_3, NL3, c_f1_3, 320, c_f2_3, 320, nullptr, 256, 256, cost3);
  float* feat3 = permf(c, (size_t)NL3 * 64);
  sfoe(c, Pf3, pc1_3, NL3, f1_3, 256, cost3, 256, nullptr, nullptr, feat3, o_flow3, o_occ3);

  // Deconv chains down the pyramid
  float* f1_3_2 = permf(c, (size_t)NL2 * 64);
  { size_t mk = c.hi; float* u = tmpf(c, (size_t)NL2 * 256);
    upsample(c, pc1_2, NL2, pc1_3, NL3, f1_3, 256, u);
    dense(c, u, NL2, Pd32, 256, 64, f1_3_2, ACT_LRELU); c.hi = mk; }
  float* f2_3_2 = permf(c, (size_t)NL2 * 64);
  { size_t mk = c.hi; float* u = tmpf(c, (size_t)NL2 * 256);
    upsample(c, pc2_2, NL2, pc2_3, NL3, f2_3, 256, u);
    dense(c, u, NL2, Pd32, 256, 64, f2_3_2, ACT_LRELU); c.hi = mk; }
  float* c_f1_2 = permf(c, (size_t)NL2 * 192);
  k_concat4<<<gE((size_t)NL2 * 192), 256, 0, stream>>>(f1_2, 128, f1_3_2, 64, nullptr, 0, nullptr, 0, c_f1_2, NL2);
  float* c_f2_2 = permf(c, (size_t)NL2 * 192);
  k_concat4<<<gE((size_t)NL2 * 192), 256, 0, stream>>>(f2_2, 128, f2_3_2, 64, nullptr, 0, nullptr, 0, c_f2_2, NL2);

  float* f1_2_1 = permf(c, (size_t)NL1 * 32);
  { size_t mk = c.hi; float* u = tmpf(c, (size_t)NL1 * 128);
    upsample(c, pc1_1, NL1, pc1_2, NL2, f1_2, 128, u);
    dense(c, u, NL1, Pd21, 128, 32, f1_2_1, ACT_LRELU); c.hi = mk; }
  float* f2_2_1 = permf(c, (size_t)NL1 * 32);
  { size_t mk = c.hi; float* u = tmpf(c, (size_t)NL1 * 128);
    upsample(c, pc2_1, NL1, pc2_2, NL2, f2_2, 128, u);
    dense(c, u, NL1, Pd21, 128, 32, f2_2_1, ACT_LRELU); c.hi = mk; }
  float* c_f1_1 = permf(c, (size_t)NL1 * 96);
  k_concat4<<<gE((size_t)NL1 * 96), 256, 0, stream>>>(f1_1, 64, f1_2_1, 32, nullptr, 0, nullptr, 0, c_f1_1, NL1);
  float* c_f2_1 = permf(c, (size_t)NL1 * 96);
  k_concat4<<<gE((size_t)NL1 * 96), 256, 0, stream>>>(f2_1, 64, f2_2_1, 32, nullptr, 0, nullptr, 0, c_f2_1, NL1);

  float* f1_1_0 = permf(c, (size_t)NL0 * 32);
  { size_t mk = c.hi; float* u = tmpf(c, (size_t)NL0 * 64);
    upsample(c, xyz1, NL0, pc1_1, NL1, f1_1, 64, u);
    dense(c, u, NL0, Pd10, 64, 32, f1_1_0, ACT_LRELU); c.hi = mk; }
  float* f2_1_0 = permf(c, (size_t)NL0 * 32);
  { size_t mk = c.hi; float* u = tmpf(c, (size_t)NL0 * 64);
    upsample(c, xyz2, NL0, pc2_1, NL1, f2_1, 64, u);
    dense(c, u, NL0, Pd10, 64, 32, f2_1_0, ACT_LRELU); c.hi = mk; }
  float* c_f1_0 = permf(c, (size_t)NL0 * 64);
  k_concat4<<<gE((size_t)NL0 * 64), 256, 0, stream>>>(f1_0, 32, f1_1_0, 32, nullptr, 0, nullptr, 0, c_f1_0, NL0);
  float* c_f2_0 = permf(c, (size_t)NL0 * 64);
  k_concat4<<<gE((size_t)NL0 * 64), 256, 0, stream>>>(f2_0, 32, f2_1_0, 32, nullptr, 0, nullptr, 0, c_f2_0, NL0);

  // Level-2 refinement
  float* up_flow2 = permf(c, (size_t)NL2 * 3);
  upsample(c, pc1_2, NL2, pc1_3, NL3, o_flow3, 3, up_flow2);
  float* up_occ2 = permf(c, (size_t)NL2);
  upsample(c, pc1_2, NL2, pc1_3, NL3, o_occ3, 1, up_occ2);
  float* pc2_2_w = permf(c, (size_t)NL2 * 3);
  warp(c, pc1_2, NL2, pc2_2, NL2, up_flow2, pc2_2_w);
  float* cost2 = permf(c, (size_t)NL2 * 128);
  mask_cv(c, Pcost2, pc1_2, NL2, pc2_2_w, NL2, c_f1_2, 192, c_f2_2, 192, up_occ2, 128, 128, cost2);
  float* feat3_up = permf(c, (size_t)NL2 * 64);
  upsample(c, pc1_2, NL2, pc1_3, NL3, feat3, 64, feat3_up);
  float* feat2 = permf(c, (size_t)NL2 * 64);
  { size_t mk = c.hi; float* fx = tmpf(c, (size_t)NL2 * 192);
    k_concat4<<<gE((size_t)NL2 * 192), 256, 0, stream>>>(f1_2, 128, feat3_up, 64, nullptr, 0, nullptr, 0, fx, NL2);
    sfoe(c, Pf2, pc1_2, NL2, fx, 192, cost2, 128, up_flow2, up_occ2, feat2, o_flow2, o_occ2);
    c.hi = mk; }

  // Level-1 refinement
  float* up_flow1 = permf(c, (size_t)NL1 * 3);
  upsample(c, pc1_1, NL1, pc1_2, NL2, o_flow2, 3, up_flow1);
  float* up_occ1 = permf(c, (size_t)NL1);
  upsample(c, pc1_1, NL1, pc1_2, NL2, o_occ2, 1, up_occ1);
  float* pc2_1_w = permf(c, (size_t)NL1 * 3);
  warp(c, pc1_1, NL1, pc2_1, NL1, up_flow1, pc2_1_w);
  float* cost1 = permf(c, (size_t)NL1 * 64);
  mask_cv(c, Pcost1, pc1_1, NL1, pc2_1_w, NL1, c_f1_1, 96, c_f2_1, 96, up_occ1, 64, 64, cost1);
  float* feat2_up = permf(c, (size_t)NL1 * 64);
  upsample(c, pc1_1, NL1, pc1_2, NL2, feat2, 64, feat2_up);
  float* feat1 = permf(c, (size_t)NL1 * 64);
  { size_t mk = c.hi; float* fx = tmpf(c, (size_t)NL1 * 128);
    k_concat4<<<gE((size_t)NL1 * 128), 256, 0, stream>>>(f1_1, 64, feat2_up, 64, nullptr, 0, nullptr, 0, fx, NL1);
    sfoe(c, Pf1, pc1_1, NL1, fx, 128, cost1, 64, up_flow1, up_occ1, feat1, o_flow1, o_occ1);
    c.hi = mk; }

  // Level-0 refinement
  float* up_flow0 = permf(c, (size_t)NL0 * 3);
  upsample(c, xyz1, NL0, pc1_1, NL1, o_flow1, 3, up_flow0);
  float* up_occ0 = permf(c, (size_t)NL0);
  upsample(c, xyz1, NL0, pc1_1, NL1, o_occ1, 1, up_occ0);
  float* pc2_0_w = permf(c, (size_t)NL0 * 3);
  warp(c, xyz1, NL0, xyz2, NL0, up_flow0, pc2_0_w);
  float* cost0 = permf(c, (size_t)NL0 * 32);
  mask_cv(c, Pcost0, xyz1, NL0, pc2_0_w, NL0, c_f1_0, 64, c_f2_0, 64, up_occ0, 32, 32, cost0);
  float* feat1_up = permf(c, (size_t)NL0 * 64);
  upsample(c, xyz1, NL0, pc1_1, NL1, feat1, 64, feat1_up);
  { size_t mk = c.hi;
    float* fx = tmpf(c, (size_t)NL0 * 96);
    k_concat4<<<gE((size_t)NL0 * 96), 256, 0, stream>>>(f1_0, 32, feat1_up, 64, nullptr, 0, nullptr, 0, fx, NL0);
    float* feat0 = tmpf(c, (size_t)NL0 * 64);
    sfoe(c, Pf0, xyz1, NL0, fx, 96, cost0, 32, up_flow0, up_occ0, feat0, o_flow0, o_occ0);
    c.hi = mk; }
}
